// GE2ELoss_1726576854887
// MI455X (gfx1250) — matrix operand order repaired and checked by reading, not compiled, not run on hardware
//
#include <hip/hip_runtime.h>
#include <math.h>

// ---------------------------------------------------------------------------
// GE2E loss, fused bf16-WMMA GEMM + online log-softmax for MI455X (gfx1250).
// N=1024 speakers, M=32 utt, D=256 dims, NM=32768 rows (fixed by reference).
// sim = ef @ c.T computed as (e_hi+e_lo)(c_hi+c_lo) dropping lo*lo
// => 3x v_wmma_f32_16x16x32_bf16 per 16x16x32 step, ~fp32 accuracy.
// Row softmax (1024 logits) is streamed flash-style: never materialize sim.
// ---------------------------------------------------------------------------

#define NSPK  1024
#define MUTT  32
#define DDIM  256
#define NROWS (NSPK * MUTT)   // 32768
#define NTILES_COL (NSPK / 16)  // 64
#define ROWS_PER_BLOCK 128
#define NBLOCKS_MAIN (NROWS / ROWS_PER_BLOCK) // 256
#define NPARTIALS (NBLOCKS_MAIN * 8)          // 2048

typedef __attribute__((ext_vector_type(8)))  float  v8f;
typedef __attribute__((ext_vector_type(8)))  __bf16 v8bf;
typedef __attribute__((ext_vector_type(16))) __bf16 v16bf;

// ---- bf16 helpers (round-to-nearest-even split) ---------------------------
__device__ __forceinline__ unsigned short f2bf(float f) {
    unsigned int u = __float_as_uint(f);
    unsigned int r = u + 0x7FFFu + ((u >> 16) & 1u);
    return (unsigned short)(r >> 16);
}
__device__ __forceinline__ float bf2f(unsigned short h) {
    return __uint_as_float(((unsigned int)h) << 16);
}

// ---- 16-lane butterfly reductions (rows live in 16-lane half-waves) -------
__device__ __forceinline__ float bsum16(float v) {
    v += __shfl_xor(v, 1, 32);
    v += __shfl_xor(v, 2, 32);
    v += __shfl_xor(v, 4, 32);
    v += __shfl_xor(v, 8, 32);
    return v;
}
__device__ __forceinline__ float bmax16(float v) {
    v = fmaxf(v, __shfl_xor(v, 1, 32));
    v = fmaxf(v, __shfl_xor(v, 2, 32));
    v = fmaxf(v, __shfl_xor(v, 4, 32));
    v = fmaxf(v, __shfl_xor(v, 8, 32));
    return v;
}

__device__ __forceinline__ v8f wmma_bf16(v16bf a, v16bf b, v8f c) {
    return __builtin_amdgcn_wmma_f32_16x16x32_bf16(
        /*neg_a=*/false, a, /*neg_b=*/false, b,
        /*c_mod=*/(short)0, c, /*reuse_a=*/false, /*reuse_b=*/false);
}

#define SHUF16(a, b) __builtin_shufflevector((a), (b), 0,1,2,3,4,5,6,7,8,9,10,11,12,13,14,15)

// ---------------------------------------------------------------------------
// Kernel 1: centroids c[n][d] = mean_m e[n][m][d]; also bf16 hi/lo split of c
// ---------------------------------------------------------------------------
__global__ void k_centroid(const float* __restrict__ e, float* __restrict__ c,
                           unsigned short* __restrict__ chi,
                           unsigned short* __restrict__ clo) {
    const int n = blockIdx.x;
    const int d = threadIdx.x;
    const float* p = e + ((size_t)n * MUTT) * DDIM + d;
    float s = 0.f;
#pragma unroll
    for (int m = 0; m < MUTT; ++m) s += p[(size_t)m * DDIM];
    const float cv = s * (1.0f / (float)MUTT);
    const size_t idx = (size_t)n * DDIM + d;
    c[idx] = cv;
    const unsigned short h = f2bf(cv);
    chi[idx] = h;
    clo[idx] = f2bf(cv - bf2f(h));
}

// ---------------------------------------------------------------------------
// Kernel 2: per row (n,m): own = (M*<e,c_n> - <e,e>)/(M-1); bf16 split of e
// ---------------------------------------------------------------------------
__global__ void k_rowprep(const float* __restrict__ e, const float* __restrict__ c,
                          float* __restrict__ own,
                          unsigned short* __restrict__ ehi,
                          unsigned short* __restrict__ elo) {
    __shared__ float red[256];
    const int row = blockIdx.x;
    const int d = threadIdx.x;
    const int n = row >> 5;  // row / M
    const float ei = e[(size_t)row * DDIM + d];
    const float cc = c[(size_t)n * DDIM + d];
    const unsigned short h = f2bf(ei);
    ehi[(size_t)row * DDIM + d] = h;
    elo[(size_t)row * DDIM + d] = f2bf(ei - bf2f(h));

    // reduce sum(e*c)
    red[d] = ei * cc;
    __syncthreads();
    for (int o = 128; o > 0; o >>= 1) {
        if (d < o) red[d] += red[d + o];
        __syncthreads();
    }
    const float dotS = red[0] * (float)MUTT;   // <e, s_n>
    __syncthreads();
    // reduce sum(e*e)
    red[d] = ei * ei;
    __syncthreads();
    for (int o = 128; o > 0; o >>= 1) {
        if (d < o) red[d] += red[d + o];
        __syncthreads();
    }
    if (d == 0) own[row] = (dotS - red[0]) * (1.0f / (float)(MUTT - 1));
}

// ---------------------------------------------------------------------------
// Kernel 3: fused GEMM (3x bf16 WMMA split-precision) + online log-softmax.
// Block: 256 threads = 8 waves; wave w owns rows [blk*128 + w*16, +16).
// Loop over 64 column tiles; B tile (16 centroids x 256 K, hi+lo) staged in
// LDS with +8 bf16 row padding (stride 264 -> bank-conflict-free b128 reads).
// ---------------------------------------------------------------------------
__global__ void __launch_bounds__(256)
k_gemm_softmax(const unsigned short* __restrict__ ehi,
               const unsigned short* __restrict__ elo,
               const unsigned short* __restrict__ chi,
               const unsigned short* __restrict__ clo,
               const float* __restrict__ own,
               const float* __restrict__ wp,
               const float* __restrict__ bp,
               float* __restrict__ partials) {
    __shared__ __align__(16) unsigned short sBhi[16 * 264];
    __shared__ __align__(16) unsigned short sBlo[16 * 264];

    const int tid   = threadIdx.x;
    const int lane  = tid & 31;
    const int wv    = tid >> 5;
    const int l16   = lane & 15;
    const int khalf = lane >> 4;
    const int rowBase = blockIdx.x * ROWS_PER_BLOCK + wv * 16;
    const float w = *wp;
    const float b = *bp;

    // ---- resident A fragments: 8 k-steps of 16x32 bf16, hi and lo --------
    // A layout (16-bit 16x32): lanes 0-15 -> K {k0..k0+7, k0+16..k0+23},
    // lanes 16-31 -> K {k0+8.., k0+24..}; per-lane v16bf = [8 @ k0+khalf*8,
    // 8 @ k0+16+khalf*8].
    v16bf Ahi[8], Alo[8];
    {
        const unsigned short* ah = ehi + (size_t)(rowBase + l16) * DDIM;
        const unsigned short* al = elo + (size_t)(rowBase + l16) * DDIM;
#pragma unroll
        for (int ks = 0; ks < 8; ++ks) {
            const int k0 = ks * 32 + khalf * 8;
            v8bf h0 = *(const v8bf*)(ah + k0);
            v8bf h1 = *(const v8bf*)(ah + k0 + 16);
            Ahi[ks] = SHUF16(h0, h1);
            v8bf l0 = *(const v8bf*)(al + k0);
            v8bf l1 = *(const v8bf*)(al + k0 + 16);
            Alo[ks] = SHUF16(l0, l1);
        }
    }

    // rows of this tile all share one speaker label (rowBase is 16-aligned,
    // M=32): n = rowBase/32; label column = n -> tile n/16, local n%16.
    const int nlab   = rowBase >> 5;
    const int ctlab  = nlab >> 4;
    const int collab = nlab & 15;
    float ownv[8];
#pragma unroll
    for (int r = 0; r < 8; ++r) ownv[r] = own[rowBase + khalf * 8 + r];

    float m_[8], s_[8], lab[8];
#pragma unroll
    for (int r = 0; r < 8; ++r) { m_[r] = -1.0e30f; s_[r] = 0.f; lab[r] = 0.f; }

    for (int ct = 0; ct < NTILES_COL; ++ct) {
        __syncthreads();
        // cooperative stage of B tile: 16 cols x 256 K bf16 (hi+lo), 16B chunks
        for (int i = tid; i < 512; i += 256) {
            const int col = i >> 5;       // 32 chunks of 8 bf16 per column
            const int cc  = i & 31;
            const size_t g = (size_t)(ct * 16 + col) * DDIM + cc * 8;
            *(v8bf*)&sBhi[col * 264 + cc * 8] = *(const v8bf*)(chi + g);
            *(v8bf*)&sBlo[col * 264 + cc * 8] = *(const v8bf*)(clo + g);
        }
        __syncthreads();

        v8f acc = {0.f, 0.f, 0.f, 0.f, 0.f, 0.f, 0.f, 0.f};
#pragma unroll
        for (int ks = 0; ks < 8; ++ks) {
            // B layout (16-bit 32x16): lanes 0-15 -> K {0..7, 16..23}, lanes 16-31 ->
            // K {8..15, 24..31}; per-lane v16bf = [8 @ khalf*8, 8 @ 16+khalf*8] of column l16.
            const int kb = ks * 32 + khalf * 8;
            const unsigned short* bh = &sBhi[l16 * 264 + kb];
            const unsigned short* bl = &sBlo[l16 * 264 + kb];
            v8bf b0 = *(const v8bf*)(bh);
            v8bf b1 = *(const v8bf*)(bh + 16);
            v16bf Bh = SHUF16(b0, b1);
            v8bf c0 = *(const v8bf*)(bl);
            v8bf c1 = *(const v8bf*)(bl + 16);
            v16bf Bl = SHUF16(c0, c1);
            acc = wmma_bf16(Ahi[ks], Bh, acc);   // hi*hi
            acc = wmma_bf16(Ahi[ks], Bl, acc);   // hi*lo
            acc = wmma_bf16(Alo[ks], Bh, acc);   // lo*hi
        }

        // online log-softmax update; C/D element r is row (r + 8*khalf),
        // column (ct*16 + l16).
        const bool labTile = (ct == ctlab);
#pragma unroll
        for (int r = 0; r < 8; ++r) {
            float logit = fmaf(w, acc[r], b);
            if (labTile && (l16 == collab)) logit = fmaf(w, ownv[r], b);
            if (labTile) {
                float t = (l16 == collab) ? logit : 0.f;
                lab[r] = bsum16(t);  // broadcast label logit to all 16 lanes
            }
            const float lmax = bmax16(logit);
            const float nm = fmaxf(m_[r], lmax);
            const float ps = bsum16(__expf(logit - nm));
            s_[r] = s_[r] * __expf(m_[r] - nm) + ps;
            m_[r] = nm;
        }
    }

    // per-row loss = logsumexp - label_logit; each row replicated over 16 lanes
    float lsum = 0.f;
#pragma unroll
    for (int r = 0; r < 8; ++r) lsum += (m_[r] + __logf(s_[r])) - lab[r];
    float v = (l16 == 0) ? lsum : 0.f;   // lanes 0 & 16 hold the two halves
    v += __shfl_xor(v, 16, 32);
    v += __shfl_xor(v, 8, 32);
    v += __shfl_xor(v, 4, 32);
    v += __shfl_xor(v, 2, 32);
    v += __shfl_xor(v, 1, 32);
    if (lane == 0) partials[blockIdx.x * 8 + wv] = v;
}

// ---------------------------------------------------------------------------
// Kernel 4: deterministic fixed-order reduction of 2048 partials -> loss
// ---------------------------------------------------------------------------
__global__ void k_finalize(const float* __restrict__ partials, float* __restrict__ out) {
    __shared__ float red[256];
    const int t = threadIdx.x;
    float s = 0.f;
    for (int i = t; i < NPARTIALS; i += 256) s += partials[i];
    red[t] = s;
    __syncthreads();
    for (int o = 128; o > 0; o >>= 1) {
        if (t < o) red[t] += red[t + o];
        __syncthreads();
    }
    if (t == 0) out[0] = red[0] * (1.0f / (float)NROWS);
}

// ---------------------------------------------------------------------------
extern "C" void kernel_launch(void* const* d_in, const int* in_sizes, int n_in,
                              void* d_out, int out_size, void* d_ws, size_t ws_size,
                              hipStream_t stream) {
    (void)in_sizes; (void)n_in; (void)out_size; (void)ws_size;
    const float* e  = (const float*)d_in[0];
    const float* wp = (const float*)d_in[1];
    const float* bp = (const float*)d_in[2];
    // d_in[3] = N, d_in[4] = M (fixed by the reference: 1024, 32)

    char* ws = (char*)d_ws;
    size_t off = 0;
    float* c        = (float*)(ws + off); off += (size_t)NSPK * DDIM * sizeof(float);
    float* own      = (float*)(ws + off); off += (size_t)NROWS * sizeof(float);
    float* partials = (float*)(ws + off); off += (size_t)NPARTIALS * sizeof(float);
    off = (off + 255) & ~(size_t)255;
    unsigned short* chi = (unsigned short*)(ws + off); off += (size_t)NSPK * DDIM * 2;
    unsigned short* clo = (unsigned short*)(ws + off); off += (size_t)NSPK * DDIM * 2;
    unsigned short* ehi = (unsigned short*)(ws + off); off += (size_t)NROWS * DDIM * 2;
    unsigned short* elo = (unsigned short*)(ws + off); off += (size_t)NROWS * DDIM * 2;
    // total workspace use: ~35.8 MB

    k_centroid<<<NSPK, 256, 0, stream>>>(e, c, chi, clo);
    k_rowprep<<<NROWS, 256, 0, stream>>>(e, c, own, ehi, elo);
    k_gemm_softmax<<<NBLOCKS_MAIN, 256, 0, stream>>>(ehi, elo, chi, clo, own, wp, bp, partials);
    k_finalize<<<1, 256, 0, stream>>>(partials, (float*)d_out);
}
